// SelfAttentionLayer_18957985645060
// MI455X (gfx1250) — matrix operand order repaired and checked by reading, not compiled, not run on hardware
//
#include <hip/hip_runtime.h>
#include <hip/hip_bf16.h>

#define B_ 16
#define T_ 2048
#define D_ 512
#define QT 32     // query rows per workgroup
#define KT 128    // keys per tile

typedef float  v8f  __attribute__((ext_vector_type(8)));
typedef __bf16 v16bf __attribute__((ext_vector_type(16)));
typedef __bf16 v8bf  __attribute__((ext_vector_type(8)));

union ABFrag { v16bf v; v8bf h[2]; };

__device__ inline v8f vzero8() { v8f z = {0.f,0.f,0.f,0.f,0.f,0.f,0.f,0.f}; return z; }

__device__ inline v8f wmma_bf16(const ABFrag& a, const ABFrag& b, v8f c) {
  return __builtin_amdgcn_wmma_f32_16x16x32_bf16(
      false, a.v, false, b.v, (short)0, c, false, false);
}

__device__ inline v8bf cvt8(float4 a, float4 b) {
  v8bf r;
  r[0]=(__bf16)a.x; r[1]=(__bf16)a.y; r[2]=(__bf16)a.z; r[3]=(__bf16)a.w;
  r[4]=(__bf16)b.x; r[5]=(__bf16)b.y; r[6]=(__bf16)b.z; r[7]=(__bf16)b.w;
  return r;
}

// CDNA5 async global->LDS copy, 16 bytes per lane.  Tracked by ASYNCcnt.
// Low 32 bits of a generic LDS pointer are the LDS byte address.
__device__ inline void async_b128(void* lds, const void* g) {
  asm volatile("global_load_async_to_lds_b128 %0, %1, off"
               :: "v"((unsigned)(uintptr_t)lds), "v"(g) : "memory");
}
__device__ inline void wait_async0() {
  asm volatile("s_wait_asynccnt 0x0" ::: "memory");
}

// ---------------------------------------------------------------------------
// Kernel 1: zero output accumulator + build bf16 transposed weights
// ---------------------------------------------------------------------------
__global__ __launch_bounds__(256) void prep_kernel(
    const float* __restrict__ Wq, const float* __restrict__ Wk,
    __bf16* __restrict__ WqT, __bf16* __restrict__ WkT,
    float* __restrict__ out) {
  int i = blockIdx.x * 256 + threadIdx.x;       // 0 .. 2*512*512-1
  if (i < B_ * D_) out[i] = 0.f;
  int which = i >> 18;                          // 512*512 = 1<<18
  int j = i & ((1 << 18) - 1);
  int r = j >> 9, c = j & (D_ - 1);             // WT[r][c] = W[c][r]
  const float* W = which ? Wk : Wq;
  __bf16* WT = which ? WkT : WqT;
  WT[(size_t)r * D_ + c] = (__bf16)W[(size_t)c * D_ + r];
}

// ---------------------------------------------------------------------------
// Kernel 1b: xbT[b][d][t] = (bf16) x[b][t][d]  (one-time value transpose;
// makes the attention value staging a contiguous async copy, L2-resident)
// ---------------------------------------------------------------------------
__global__ __launch_bounds__(256) void xT_kernel(
    const float* __restrict__ x, __bf16* __restrict__ xbT) {
  int b = blockIdx.z;
  int t0 = blockIdx.x * 32, d0 = blockIdx.y * 32;
  __shared__ float tile[32][33];
  int tr = threadIdx.x >> 3;          // 0..31
  int tc = (threadIdx.x & 7) * 4;     // 0,4..28
  const float* src = x + ((size_t)b * T_ + t0 + tr) * D_ + d0 + tc;
  float4 f = *(const float4*)src;
  tile[tr][tc+0]=f.x; tile[tr][tc+1]=f.y; tile[tr][tc+2]=f.z; tile[tr][tc+3]=f.w;
  __syncthreads();
  __bf16* dst = xbT + ((size_t)b * D_ + d0 + tr) * T_ + t0 + tc;
  dst[0]=(__bf16)tile[tc+0][tr]; dst[1]=(__bf16)tile[tc+1][tr];
  dst[2]=(__bf16)tile[tc+2][tr]; dst[3]=(__bf16)tile[tc+3][tr];
}

// ---------------------------------------------------------------------------
// Kernel 2: Q = x*Wq, K = x*Wk  (bf16 out).  One 16x16 tile per wave.
// ---------------------------------------------------------------------------
__global__ __launch_bounds__(256) void proj_kernel(
    const float* __restrict__ x,
    const __bf16* __restrict__ WqT, const __bf16* __restrict__ WkT,
    __bf16* __restrict__ Qb, __bf16* __restrict__ Kb) {
  int lane = threadIdx.x & 31;
  int wave = threadIdx.x >> 5;
  int hl = lane & 15, g = lane >> 4;

  int gt = blockIdx.x * 8 + wave;               // 0 .. 131071
  int which = gt >> 16;                         // 0 = Q, 1 = K
  int t2 = gt & 65535;
  int mtile = t2 >> 5;                          // rows of [B*T]
  int ntile = t2 & 31;                          // cols of D
  const __bf16* WT = which ? WkT : WqT;
  __bf16* out = which ? Kb : Qb;

  v8f acc = vzero8();
  const float* xr = x + (size_t)(mtile * 16 + hl) * D_;
  const __bf16* wr = WT + (size_t)(ntile * 16 + hl) * D_;

#pragma unroll 4
  for (int kc = 0; kc < D_ / 32; ++kc) {
    int d0 = kc * 32;
    ABFrag a, b;
    float4 f0 = *(const float4*)(xr + d0 + g * 8);
    float4 f1 = *(const float4*)(xr + d0 + g * 8 + 4);
    float4 f2 = *(const float4*)(xr + d0 + 16 + g * 8);
    float4 f3 = *(const float4*)(xr + d0 + 16 + g * 8 + 4);
    a.h[0] = cvt8(f0, f1);
    a.h[1] = cvt8(f2, f3);
    b.h[0] = *(const v8bf*)(wr + d0 + g * 8);
    b.h[1] = *(const v8bf*)(wr + d0 + 16 + g * 8);
    acc = wmma_bf16(a, b, acc);
  }
#pragma unroll
  for (int j = 0; j < 8; ++j)
    out[(size_t)(mtile * 16 + g * 8 + j) * D_ + ntile * 16 + hl] = (__bf16)acc[j];
}

// ---------------------------------------------------------------------------
// Kernel 3: flash attention + fused mean pool.  One WG = (batch, 32 queries).
// Async double-buffered K/value staging; 8 waves; wave w owns output cols
// [w*64, w*64+64).
// ---------------------------------------------------------------------------
__global__ __launch_bounds__(256) void attn_kernel(
    const __bf16* __restrict__ xbT,
    const __bf16* __restrict__ Qb, const __bf16* __restrict__ Kb,
    float* __restrict__ out) {
  int b = blockIdx.y;
  int q0 = blockIdx.x * QT;
  int tid = threadIdx.x;
  int lane = tid & 31, wave = tid >> 5;
  int hl = lane & 15, g = lane >> 4;

  __shared__ __align__(16) __bf16 Qs[QT][D_];        // 32 KB
  __shared__ __align__(16) __bf16 Ks[2][KT][32];     // 16 KB (dbl buf)
  __shared__ __align__(16) __bf16 XsT[2][D_][32];    // 64 KB (dbl buf)
  __shared__ __align__(16) float  Ssc[QT][KT];       // 16 KB
  __shared__ __align__(16) __bf16 Ps[QT][KT];        // 8 KB
  __shared__ float mrow[QT], lrow[QT], arow[QT];

  // --- stage Q tile via async copy (rows q0..q0+31 contiguous in Qb) ---
  {
    const __bf16* qsrc = Qb + ((size_t)b * T_ + q0) * D_;
#pragma unroll
    for (int i = 0; i < 8; ++i) {
      int f = tid + i * 256;                     // 0..2047, 16B each
      async_b128(&Qs[0][0] + f * 8, qsrc + f * 8);
    }
  }
  if (tid < QT) { mrow[tid] = -INFINITY; lrow[tid] = 0.f; }

  v8f O[2][4];
#pragma unroll
  for (int mt = 0; mt < 2; ++mt)
#pragma unroll
    for (int nt = 0; nt < 4; ++nt) O[mt][nt] = vzero8();

  wait_async0();
  __syncthreads();

#pragma unroll 1
  for (int kt = 0; kt < T_ / KT; ++kt) {
    int k0 = kt * KT;
    const __bf16* kbase = Kb + ((size_t)b * T_ + k0) * D_;
    const __bf16* xbase = xbT + (size_t)b * D_ * T_ + k0;

    // ---- S = Q * Ktile^T over 16 d-chunks, double-buffered async K ----
    v8f Sacc[2];
    Sacc[0] = vzero8(); Sacc[1] = vzero8();
    {  // issue chunk 0
#pragma unroll
      for (int i = 0; i < 2; ++i) {
        int f = tid + i * 256;                   // 0..511
        int row = f >> 2, part = f & 3;
        async_b128(&Ks[0][row][part * 8], kbase + (size_t)row * D_ + part * 8);
      }
    }
    for (int dc = 0; dc < D_ / 32; ++dc) {
      int cur = dc & 1;
      wait_async0();
      __syncthreads();
      if (dc + 1 < D_ / 32) {                    // prefetch next chunk
#pragma unroll
        for (int i = 0; i < 2; ++i) {
          int f = tid + i * 256;
          int row = f >> 2, part = f & 3;
          async_b128(&Ks[1 - cur][row][part * 8],
                     kbase + (size_t)row * D_ + (dc + 1) * 32 + part * 8);
        }
      }
      int d0 = dc * 32;
#pragma unroll
      for (int mt = 0; mt < 2; ++mt) {
        ABFrag a, kb;
        int row = mt * 16 + hl;
        a.h[0] = *(const v8bf*)&Qs[row][d0 + g * 8];
        a.h[1] = *(const v8bf*)&Qs[row][d0 + 16 + g * 8];
        int key = wave * 16 + hl;
        kb.h[0] = *(const v8bf*)&Ks[cur][key][g * 8];
        kb.h[1] = *(const v8bf*)&Ks[cur][key][16 + g * 8];
        Sacc[mt] = wmma_bf16(a, kb, Sacc[mt]);
      }
    }

    // issue value chunk 0 early: overlaps spill + softmax
#pragma unroll
    for (int i = 0; i < 8; ++i) {
      int f = tid + i * 256;                     // 0..2047
      int d = f >> 2, part = f & 3;
      async_b128(&XsT[0][d][part * 8], xbase + (size_t)d * T_ + part * 8);
    }

    // spill S (C layout: VGPR j -> row mt*16+g*8+j, col wave*16+hl)
#pragma unroll
    for (int mt = 0; mt < 2; ++mt)
#pragma unroll
      for (int j = 0; j < 8; ++j)
        Ssc[mt * 16 + g * 8 + j][wave * 16 + hl] = Sacc[mt][j];
    __syncthreads();

    // ---- online softmax: 8 threads per row (rows stay within a wave) ----
    {
      int row = tid >> 3, seg = tid & 7;
      float vmax = -INFINITY;
#pragma unroll
      for (int c = 0; c < 16; ++c) vmax = fmaxf(vmax, Ssc[row][seg * 16 + c]);
#pragma unroll
      for (int o = 4; o >= 1; o >>= 1) vmax = fmaxf(vmax, __shfl_xor(vmax, o, 32));
      float mold = mrow[row];
      float mnew = fmaxf(mold, vmax);
      float alpha = __expf(mold - mnew);
      float psum = 0.f;
#pragma unroll
      for (int c = 0; c < 16; ++c) {
        float p = __expf(Ssc[row][seg * 16 + c] - mnew);
        Ps[row][seg * 16 + c] = (__bf16)p;
        psum += p;
      }
#pragma unroll
      for (int o = 4; o >= 1; o >>= 1) psum += __shfl_xor(psum, o, 32);
      if (seg == 0) {
        mrow[row] = mnew;
        lrow[row] = lrow[row] * alpha + psum;
        arow[row] = alpha;
      }
    }
    __syncthreads();

    // ---- rescale O accumulators by per-row alpha ----
#pragma unroll
    for (int mt = 0; mt < 2; ++mt)
#pragma unroll
      for (int j = 0; j < 8; ++j) {
        float al = arow[mt * 16 + g * 8 + j];
#pragma unroll
        for (int nt = 0; nt < 4; ++nt) O[mt][nt][j] *= al;
      }

    // ---- O += P * V, 4 key chunks of 32, double-buffered async values ----
    for (int kc = 0; kc < KT / 32; ++kc) {
      int cur = kc & 1;
      wait_async0();
      __syncthreads();
      if (kc + 1 < KT / 32) {                    // prefetch next value chunk
#pragma unroll
        for (int i = 0; i < 8; ++i) {
          int f = tid + i * 256;
          int d = f >> 2, part = f & 3;
          async_b128(&XsT[1 - cur][d][part * 8],
                     xbase + (size_t)d * T_ + (kc + 1) * 32 + part * 8);
        }
      }
#pragma unroll
      for (int mt = 0; mt < 2; ++mt) {
        ABFrag a;
        int row = mt * 16 + hl;
        a.h[0] = *(const v8bf*)&Ps[row][kc * 32 + g * 8];
        a.h[1] = *(const v8bf*)&Ps[row][kc * 32 + 16 + g * 8];
#pragma unroll
        for (int nt = 0; nt < 4; ++nt) {
          ABFrag xb;
          int dcol = wave * 64 + nt * 16 + hl;
          xb.h[0] = *(const v8bf*)&XsT[cur][dcol][g * 8];
          xb.h[1] = *(const v8bf*)&XsT[cur][dcol][16 + g * 8];
          O[mt][nt] = wmma_bf16(a, xb, O[mt][nt]);
        }
      }
    }
  }

  // ---- fused mean pool: out[b][d] += sum_rows O[row][d] / (l[row]*T) ----
  __syncthreads();
  if (tid < QT) lrow[tid] = 1.0f / (lrow[tid] * (float)T_);
  __syncthreads();
#pragma unroll
  for (int nt = 0; nt < 4; ++nt) {
    float s = 0.f;
#pragma unroll
    for (int mt = 0; mt < 2; ++mt)
#pragma unroll
      for (int j = 0; j < 8; ++j)
        s += O[mt][nt][j] * lrow[mt * 16 + g * 8 + j];
    s += __shfl_xor(s, 16, 32);           // combine row halves (same column)
    if (g == 0)
      atomicAdd(&out[b * D_ + wave * 64 + nt * 16 + hl], s);
  }
}

// ---------------------------------------------------------------------------
extern "C" void kernel_launch(void* const* d_in, const int* in_sizes, int n_in,
                              void* d_out, int out_size, void* d_ws, size_t ws_size,
                              hipStream_t stream) {
  const float* x  = (const float*)d_in[0];
  const float* Wq = (const float*)d_in[1];
  const float* Wk = (const float*)d_in[2];
  float* out = (float*)d_out;

  // ws (bf16): Q[B*T*D] | K[B*T*D] | WqT[D*D] | WkT[D*D] | xbT[B*D*T]
  __bf16* Qb  = (__bf16*)d_ws;
  __bf16* Kb  = Qb  + (size_t)B_ * T_ * D_;
  __bf16* WqT = Kb  + (size_t)B_ * T_ * D_;
  __bf16* WkT = WqT + (size_t)D_ * D_;
  __bf16* xbT = WkT + (size_t)D_ * D_;

  prep_kernel<<<(2 * D_ * D_) / 256, 256, 0, stream>>>(Wq, Wk, WqT, WkT, out);
  xT_kernel<<<dim3(T_ / 32, D_ / 32, B_), 256, 0, stream>>>(x, xbT);
  proj_kernel<<<(2 * (B_ * T_ / 16) * (D_ / 16)) / 8, 256, 0, stream>>>(x, WqT, WkT, Qb, Kb);
  attn_kernel<<<dim3(T_ / QT, B_), 256, 0, stream>>>(xbT, Qb, Kb, out);
}
